// PerformerAttention_7078106103917
// MI455X (gfx1250) — matrix operand order repaired and checked by reading, not compiled, not run on hardware
//
#include <hip/hip_runtime.h>

#define B_DIM 4
#define H_DIM 8
#define N_DIM 4096
#define D_DIM 64
#define M_DIM 256

constexpr float SCALE = 0.35355339059327379f;  // 64^-0.25
constexpr float RATIO = 0.0625f;               // 256^-0.5
constexpr float EPSV  = 1.0e-4f;
#define NEG_BIG (-3.0e38f)

typedef __bf16 bf16;
typedef __attribute__((ext_vector_type(16))) __bf16 v16bf;
typedef __attribute__((ext_vector_type(8)))  __bf16 v8bf;
typedef __attribute__((ext_vector_type(8)))  float  v8f;

static __device__ __forceinline__ v8f wmma_bf16(v16bf a, v16bf b, v8f c) {
  // D = A(16x32 bf16) * B(32x16 bf16) + C(16x16 f32)
  return __builtin_amdgcn_wmma_f32_16x16x32_bf16(false, a, false, b, (short)0, c, false, false);
}

// A-matrix fragment: LDS layout A[row][k], k contiguous, 'stride' elements per row.
// ISA 16-bit A 16x32: lanes 0-15 -> K 0..7 / 16..23 ; lanes 16-31 -> K 8..15 / 24..31
static __device__ __forceinline__ v16bf load_a_frag(const bf16* A, int stride, int r0, int k0, int lane) {
  const int hh = (lane >> 4) & 1;
  const bf16* p = A + (r0 + (lane & 15)) * stride + k0 + hh * 8;
  v8bf lo = *(const v8bf*)(p);
  v8bf hi = *(const v8bf*)(p + 16);
  v16bf out;
#pragma unroll
  for (int i = 0; i < 8; ++i) { out[i] = lo[i]; out[i + 8] = hi[i]; }
  return out;
}

// B-matrix fragment from transposed staging Bt[n][k] (k contiguous):
// lanes 0-15 -> K 0..7 / 16..23 of column n0+lane ; lanes 16-31 -> K 8..15 / 24..31
static __device__ __forceinline__ v16bf load_b_frag(const bf16* Bt, int stride, int n0, int k0, int lane) {
  const int hh = (lane >> 4) & 1;
  const bf16* p = Bt + (n0 + (lane & 15)) * stride + k0 + hh * 8;
  v8bf lo = *(const v8bf*)(p);
  v8bf hi = *(const v8bf*)(p + 16);
  v16bf out;
#pragma unroll
  for (int i = 0; i < 8; ++i) { out[i] = lo[i]; out[i + 8] = hi[i]; }
  return out;
}

__global__ void zero_kernel(float* __restrict__ p, int n) {
  int i = blockIdx.x * blockDim.x + threadIdx.x;
  if (i < n) p[i] = 0.0f;
}

// ---------------- Kernel A: per-(bh,chunk) max of xp_k -------------------
// grid (8 chunks, 32 heads), 256 threads (8 waves)
__global__ void stab_partial_kernel(const float* __restrict__ Kg,
                                    const float* __restrict__ maskg,
                                    const float* __restrict__ projg,
                                    float* __restrict__ stabp) {
  extern __shared__ __attribute__((aligned(64))) char smem[];
  bf16*  projT = (bf16*)(smem);            // 256*64 bf16 = 32768 B
  bf16*  Abuf  = (bf16*)(smem + 32768);    // 64*64 bf16  = 8192 B
  float* redS  = (float*)(smem + 40960);   // 256 f32     = 1024 B

  const int tid = threadIdx.x, lane = tid & 31, wave = tid >> 5;
  const int chunk = blockIdx.x, bh = blockIdx.y, b = bh >> 3;

  for (int i = tid; i < M_DIM * D_DIM; i += 256) projT[i] = (bf16)projg[i];
  __syncthreads();

  v16bf pb[2][2];
#pragma unroll
  for (int ci = 0; ci < 2; ++ci)
#pragma unroll
    for (int kk = 0; kk < 2; ++kk)
      pb[ci][kk] = load_b_frag(projT, D_DIM, (wave * 2 + ci) * 16, kk * 32, lane);

  const int row = tid >> 2, part = tid & 3;
  float mx = NEG_BIG;

  for (int it = 0; it < 8; ++it) {
    const int n0 = chunk * 512 + it * 64;
    __syncthreads();
    {
      const int n = n0 + row;
      const float mval = maskg[b * N_DIM + n] * SCALE;
      const float* kr = Kg + ((size_t)bh * N_DIM + n) * D_DIM + part * 16;
#pragma unroll
      for (int j = 0; j < 16; ++j)
        Abuf[row * 64 + part * 16 + j] = (bf16)(kr[j] * mval);
    }
    __syncthreads();
#pragma unroll
    for (int r = 0; r < 4; ++r) {
      v16bf a0 = load_a_frag(Abuf, 64, r * 16, 0, lane);
      v16bf a1 = load_a_frag(Abuf, 64, r * 16, 32, lane);
#pragma unroll
      for (int ci = 0; ci < 2; ++ci) {
        v8f acc = {0.f, 0.f, 0.f, 0.f, 0.f, 0.f, 0.f, 0.f};
        acc = wmma_bf16(a0, pb[ci][0], acc);
        acc = wmma_bf16(a1, pb[ci][1], acc);
#pragma unroll
        for (int v = 0; v < 8; ++v) mx = fmaxf(mx, acc[v]);
      }
    }
  }
  __syncthreads();
  redS[tid] = mx;
  __syncthreads();
  for (int s = 128; s > 0; s >>= 1) {
    if (tid < s) redS[tid] = fmaxf(redS[tid], redS[tid + s]);
    __syncthreads();
  }
  if (tid == 0) stabp[bh * 8 + chunk] = redS[0];
}

__global__ void stab_reduce_kernel(const float* __restrict__ sp, float* __restrict__ st) {
  int t = threadIdx.x;
  if (t < 32) {
    float m = NEG_BIG;
    for (int c = 0; c < 8; ++c) m = fmaxf(m, sp[t * 8 + c]);
    st[t] = m;
  }
}

// ---------------- Kernel B: context = phi_k^T v, k_sum ------------------
// grid (8 chunks, 32 heads), 256 threads
__global__ void context_kernel(const float* __restrict__ Kg,
                               const float* __restrict__ Vg,
                               const float* __restrict__ maskg,
                               const float* __restrict__ projg,
                               const float* __restrict__ stabg,
                               float* __restrict__ ctxg,
                               float* __restrict__ ksumg) {
  extern __shared__ __attribute__((aligned(64))) char smem[];
  bf16*  projT = (bf16*)(smem);            // 32768 B
  bf16*  Abuf  = (bf16*)(smem + 32768);    // 8192 B  k-tile [row][d]
  bf16*  vT    = (bf16*)(smem + 40960);    // 8192 B  v transposed [e][row]
  bf16*  phiT  = (bf16*)(smem + 49152);    // 32768 B phi_k transposed [m][row]
  float* diagS = (float*)(smem + 81920);   // 256 B
  float* redS  = (float*)(smem + 82176);   // 1024 B

  const int tid = threadIdx.x, lane = tid & 31, wave = tid >> 5;
  const int hh = (lane >> 4) & 1;
  const int chunk = blockIdx.x, bh = blockIdx.y, b = bh >> 3;
  const float stab = stabg[bh];

  for (int i = tid; i < M_DIM * D_DIM; i += 256) projT[i] = (bf16)projg[i];
  __syncthreads();

  v16bf pb[2][2];
#pragma unroll
  for (int ci = 0; ci < 2; ++ci)
#pragma unroll
    for (int kk = 0; kk < 2; ++kk)
      pb[ci][kk] = load_b_frag(projT, D_DIM, (wave * 2 + ci) * 16, kk * 32, lane);

  const int row = tid >> 2, part = tid & 3;
  float ksacc[2] = {0.f, 0.f};
  v8f acc2[2][4];
#pragma unroll
  for (int mi = 0; mi < 2; ++mi)
#pragma unroll
    for (int e = 0; e < 4; ++e)
      acc2[mi][e] = (v8f){0.f, 0.f, 0.f, 0.f, 0.f, 0.f, 0.f, 0.f};

  for (int it = 0; it < 8; ++it) {
    const int n0 = chunk * 512 + it * 64;
    __syncthreads();
    {
      const int n = n0 + row;
      const float mraw = maskg[b * N_DIM + n];
      const float mval = mraw * SCALE;
      const float* kr = Kg + ((size_t)bh * N_DIM + n) * D_DIM + part * 16;
      const float* vr = Vg + ((size_t)bh * N_DIM + n) * D_DIM + part * 16;
      float ss = 0.f;
#pragma unroll
      for (int j = 0; j < 16; ++j) {
        float x = kr[j] * mval;
        ss += x * x;
        Abuf[row * 64 + part * 16 + j] = (bf16)x;
      }
#pragma unroll
      for (int j = 0; j < 16; ++j)
        vT[(part * 16 + j) * 64 + row] = (bf16)(vr[j] * mraw);
      redS[row * 4 + part] = ss;
    }
    __syncthreads();
    if (part == 0)
      diagS[row] = 0.5f * (redS[row * 4] + redS[row * 4 + 1] + redS[row * 4 + 2] + redS[row * 4 + 3]);
    __syncthreads();

    // GEMM1 (xp_k) fused with exp -> phiT[m][row] (wave-exclusive m range)
#pragma unroll
    for (int r = 0; r < 4; ++r) {
      v16bf a0 = load_a_frag(Abuf, 64, r * 16, 0, lane);
      v16bf a1 = load_a_frag(Abuf, 64, r * 16, 32, lane);
#pragma unroll
      for (int ci = 0; ci < 2; ++ci) {
        v8f acc = {0.f, 0.f, 0.f, 0.f, 0.f, 0.f, 0.f, 0.f};
        acc = wmma_bf16(a0, pb[ci][0], acc);
        acc = wmma_bf16(a1, pb[ci][1], acc);
        const int m = (wave * 2 + ci) * 16 + (lane & 15);
        const int rb = r * 16 + hh * 8;
        v8bf ph;
        float ks = 0.f;
#pragma unroll
        for (int v = 0; v < 8; ++v) {
          float p = RATIO * (__expf(acc[v] - diagS[rb + v] - stab) + EPSV);
          ks += p;
          ph[v] = (bf16)p;
        }
        ksacc[ci] += ks;
        *(v8bf*)&phiT[m * 64 + rb] = ph;
      }
    }
    __syncthreads();

    // GEMM2: context += phi_k^T (A: phiT[m][r]) x v (B: vT[e][r])
#pragma unroll
    for (int k0 = 0; k0 < 64; k0 += 32) {
      v16bf bf_[4];
#pragma unroll
      for (int e = 0; e < 4; ++e) bf_[e] = load_b_frag(vT, 64, e * 16, k0, lane);
#pragma unroll
      for (int mi = 0; mi < 2; ++mi) {
        v16bf a = load_a_frag(phiT, 64, (wave * 2 + mi) * 16, k0, lane);
#pragma unroll
        for (int e = 0; e < 4; ++e) acc2[mi][e] = wmma_bf16(a, bf_[e], acc2[mi][e]);
      }
    }
  }

  // writeback partial context / k_sum
#pragma unroll
  for (int mi = 0; mi < 2; ++mi)
#pragma unroll
    for (int e = 0; e < 4; ++e)
#pragma unroll
      for (int v = 0; v < 8; ++v) {
        int m = (wave * 2 + mi) * 16 + hh * 8 + v;
        int ee = e * 16 + (lane & 15);
        atomicAdd(&ctxg[(size_t)bh * (M_DIM * D_DIM) + m * D_DIM + ee], acc2[mi][e][v]);
      }
#pragma unroll
  for (int ci = 0; ci < 2; ++ci)
    atomicAdd(&ksumg[bh * M_DIM + (wave * 2 + ci) * 16 + (lane & 15)], ksacc[ci]);
}

// ---------------- Kernel C: out = (phi_q @ context) * D_inv -------------
// grid (64 row-tiles, 32 heads), 256 threads
__global__ void out_kernel(const float* __restrict__ Qg,
                           const float* __restrict__ projg,
                           const float* __restrict__ ctxg,
                           const float* __restrict__ ksumg,
                           float* __restrict__ outg) {
  extern __shared__ __attribute__((aligned(64))) char smem[];
  bf16*  projT = (bf16*)(smem);             // 32768
  bf16*  ctxT  = (bf16*)(smem + 32768);     // 32768  context^T [e][m]
  bf16*  phiQ  = (bf16*)(smem + 65536);     // 32768  [row][m]
  bf16*  qbuf  = (bf16*)(smem + 98304);     // 8192   [row][d]
  float* xpS   = (float*)(smem + 106496);   // 65536  [row][m]
  float* ksumS = (float*)(smem + 172032);   // 1024
  float* diagS = (float*)(smem + 173056);   // 256
  float* redS  = (float*)(smem + 173312);   // 1024
  float* red2S = (float*)(smem + 174336);   // 1024
  float* dinvS = (float*)(smem + 175360);   // 256

  const int tid = threadIdx.x, lane = tid & 31, wave = tid >> 5;
  const int hh = (lane >> 4) & 1;
  const int tile = blockIdx.x, bh = blockIdx.y;
  const int n0 = tile * 64;

  for (int i = tid; i < M_DIM * D_DIM; i += 256) projT[i] = (bf16)projg[i];
  for (int i = tid; i < D_DIM * M_DIM; i += 256) {
    int e = i >> 8, m = i & 255;
    ctxT[i] = (bf16)ctxg[(size_t)bh * (M_DIM * D_DIM) + m * D_DIM + e];
  }
  ksumS[tid] = ksumg[bh * M_DIM + tid];

  const int row = tid >> 2, part = tid & 3;
  {
    const float* qr = Qg + ((size_t)bh * N_DIM + n0 + row) * D_DIM + part * 16;
    float ss = 0.f;
#pragma unroll
    for (int j = 0; j < 16; ++j) {
      float x = qr[j] * SCALE;
      ss += x * x;
      qbuf[row * 64 + part * 16 + j] = (bf16)x;
    }
    redS[row * 4 + part] = ss;
  }
  __syncthreads();
  if (part == 0)
    diagS[row] = 0.5f * (redS[row * 4] + redS[row * 4 + 1] + redS[row * 4 + 2] + redS[row * 4 + 3]);

  v16bf pb[2][2];
#pragma unroll
  for (int ci = 0; ci < 2; ++ci)
#pragma unroll
    for (int kk = 0; kk < 2; ++kk)
      pb[ci][kk] = load_b_frag(projT, D_DIM, (wave * 2 + ci) * 16, kk * 32, lane);
  __syncthreads();

  // GEMM1: xp_q -> LDS (f32)
#pragma unroll
  for (int r = 0; r < 4; ++r) {
    v16bf a0 = load_a_frag(qbuf, 64, r * 16, 0, lane);
    v16bf a1 = load_a_frag(qbuf, 64, r * 16, 32, lane);
#pragma unroll
    for (int ci = 0; ci < 2; ++ci) {
      v8f acc = {0.f, 0.f, 0.f, 0.f, 0.f, 0.f, 0.f, 0.f};
      acc = wmma_bf16(a0, pb[ci][0], acc);
      acc = wmma_bf16(a1, pb[ci][1], acc);
      const int m = (wave * 2 + ci) * 16 + (lane & 15);
      const int rb = r * 16 + hh * 8;
#pragma unroll
      for (int v = 0; v < 8; ++v) xpS[(rb + v) * M_DIM + m] = acc[v];
    }
  }
  __syncthreads();

  // per-row max, phi_q, D_inv
  float pm = NEG_BIG;
  for (int j = 0; j < 64; ++j) pm = fmaxf(pm, xpS[row * M_DIM + part * 64 + j]);
  redS[row * 4 + part] = pm;
  __syncthreads();
  {
    float st = fmaxf(fmaxf(redS[row * 4], redS[row * 4 + 1]),
                     fmaxf(redS[row * 4 + 2], redS[row * 4 + 3]));
    float dg = diagS[row];
    float dp = 0.f;
    for (int j = 0; j < 64; ++j) {
      int m = part * 64 + j;
      float p = RATIO * (__expf(xpS[row * M_DIM + m] - dg - st) + EPSV);
      phiQ[row * M_DIM + m] = (bf16)p;
      dp += p * ksumS[m];
    }
    red2S[row * 4 + part] = dp;
  }
  __syncthreads();
  if (part == 0)
    dinvS[row] = 1.0f / (red2S[row * 4] + red2S[row * 4 + 1] + red2S[row * 4 + 2] + red2S[row * 4 + 3]);
  __syncthreads();

  // GEMM3: out = phi_q @ context, scaled by D_inv
  const int rt = wave >> 1;
  const int e0 = (wave & 1) * 32;
  v8f o0 = {0.f, 0.f, 0.f, 0.f, 0.f, 0.f, 0.f, 0.f};
  v8f o1 = {0.f, 0.f, 0.f, 0.f, 0.f, 0.f, 0.f, 0.f};
#pragma unroll
  for (int k0 = 0; k0 < M_DIM; k0 += 32) {
    v16bf a  = load_a_frag(phiQ, M_DIM, rt * 16, k0, lane);
    v16bf b0 = load_b_frag(ctxT, M_DIM, e0, k0, lane);
    v16bf b1 = load_b_frag(ctxT, M_DIM, e0 + 16, k0, lane);
    o0 = wmma_bf16(a, b0, o0);
    o1 = wmma_bf16(a, b1, o1);
  }
  const int rb = rt * 16 + hh * 8;
  const int ec0 = e0 + (lane & 15);
  const int ec1 = e0 + 16 + (lane & 15);
#pragma unroll
  for (int v = 0; v < 8; ++v) {
    float di = dinvS[rb + v];
    size_t base = ((size_t)bh * N_DIM + n0 + rb + v) * D_DIM;
    outg[base + ec0] = o0[v] * di;
    outg[base + ec1] = o1[v] * di;
  }
}

extern "C" void kernel_launch(void* const* d_in, const int* in_sizes, int n_in,
                              void* d_out, int out_size, void* d_ws, size_t ws_size,
                              hipStream_t stream) {
  const float* Q    = (const float*)d_in[0];
  const float* K    = (const float*)d_in[1];
  const float* V    = (const float*)d_in[2];
  const float* mask = (const float*)d_in[3];
  const float* proj = (const float*)d_in[4];
  float* out = (float*)d_out;

  float* ws    = (float*)d_ws;
  float* ctx   = ws;                 // 32*256*64 = 524288 floats
  float* ksum  = ws + 524288;        // 32*256    = 8192 floats
  float* stabp = ws + 532480;        // 32*8      = 256 floats
  float* stab  = ws + 532736;        // 32 floats

  const int nzero = 524288 + 8192;   // ctx + ksum
  zero_kernel<<<dim3((nzero + 255) / 256), dim3(256), 0, stream>>>(ws, nzero);

  const size_t smemA = 32768 + 8192 + 1024;
  stab_partial_kernel<<<dim3(8, 32), dim3(256), smemA, stream>>>(K, mask, proj, stabp);

  stab_reduce_kernel<<<dim3(1), dim3(32), 0, stream>>>(stabp, stab);

  const size_t smemB = 32768 + 8192 + 8192 + 32768 + 256 + 1024;  // 83200
  context_kernel<<<dim3(8, 32), dim3(256), smemB, stream>>>(K, V, mask, proj, stab, ctx, ksum);

  const size_t smemC = 175616;
  out_kernel<<<dim3(64, 32), dim3(256), smemC, stream>>>(Q, proj, ctx, ksum, out);
}
